// rotatory_decoder_block_49400713838918
// MI455X (gfx1250) — matrix operand order repaired and checked by reading, not compiled, not run on hardware
//
#include <hip/hip_runtime.h>
#include <math.h>

#define EPSF 1e-5f

typedef __attribute__((ext_vector_type(16))) _Float16 v16h;
typedef __attribute__((ext_vector_type(8)))  _Float16 v8h;
typedef __attribute__((ext_vector_type(4)))  _Float16 v4h;
typedef __attribute__((ext_vector_type(8)))  float    v8f;

// problem dims
#define NN_   64
#define INC_  256
#define OUTC_ 128
#define S_    1024   // 32*32
#define H2_   64
#define W2_   64
#define HW2_  4096   // 64*64

// GEMM tile config: 256 threads = 8 waves; each wave owns 16 rows x 64 cols (4 wmma tiles)
#define BM 128
#define BN 64
#define BK 32

// ---------------------------------------------------------------------------
// Generic f16-WMMA GEMM: C = alpha * op(A) @ op(B) + beta*C (+bias)
//   logical A[m][k] = TA ? A[k*lda+m] : A[m*lda+k]
//   logical B[k][n] = TB ? B[n*ldb+k] : B[k*ldb+n]
//   bias_mode: 0 none, 1 per-row (bias[m]), 2 per-col (bias[n])
//   EPI: 0 plain store; 1 ConvTranspose2d(2x2,s2) scatter (m=(o,p,q), n=(h,w) 32x32)
//   batch via blockIdx.z with strides sA/sB/sC
// REQUIRES: M % 128 == 0, N % 64 == 0, K % 32 == 0, lda/ldb % 4 == 0,
//           16B-aligned base pointers (all launches below satisfy this).
// ---------------------------------------------------------------------------
template<int TA, int TB, int EPI>
__global__ __launch_bounds__(256)
void gemm_wmma(const float* __restrict__ Ag, const float* __restrict__ Bg,
               float* __restrict__ Cg, const float* __restrict__ bias,
               int M, int N, int K, int lda, int ldb, int ldc,
               long long sA, long long sB, long long sC,
               float alpha, float beta, int bias_mode)
{
    __shared__ __align__(32) _Float16 As[BM][BK];
    __shared__ __align__(32) _Float16 Bs[BN][BK];

    const int tid   = threadIdx.x;
    const int wave  = tid >> 5;
    const int lane  = tid & 31;
    const int row16 = lane & 15;
    const int half  = lane >> 4;

    const int m0 = blockIdx.y * BM;
    const int n0 = blockIdx.x * BN;
    const long long zb = blockIdx.z;
    const float* A = Ag + zb * sA;
    const float* B = Bg + zb * sB;
    float*       C = Cg + zb * sC;

    v8f acc[4];
#pragma unroll
    for (int t = 0; t < 4; ++t)
#pragma unroll
        for (int e = 0; e < 8; ++e) acc[t][e] = 0.0f;

    for (int k0 = 0; k0 < K; k0 += BK) {
        if (tid == 0 && k0 + BK < K) {
            __builtin_prefetch(TA ? &A[(long long)(k0 + BK) * lda + m0]
                                  : &A[(long long)m0 * lda + (k0 + BK)], 0, 3);
            __builtin_prefetch(TB ? &B[(long long)n0 * ldb + (k0 + BK)]
                                  : &B[(long long)(k0 + BK) * ldb + n0], 0, 3);
        }

        // ---- stage A tile (BM x BK), f32 -> f16, float4 global loads ----
        if (TA == 0) {
            // memory contiguous along k: thread owns 4 consecutive k
#pragma unroll
            for (int it = 0; it < 4; ++it) {
                int mm = (tid >> 3) + it * 32;
                int kc = (tid & 7) * 4;
                float4 v = *(const float4*)&A[(long long)(m0 + mm) * lda + (k0 + kc)];
                v4h h; h[0] = (_Float16)v.x; h[1] = (_Float16)v.y;
                       h[2] = (_Float16)v.z; h[3] = (_Float16)v.w;
                *(v4h*)&As[mm][kc] = h;   // single ds_store_b64
            }
        } else {
            // memory contiguous along m: thread owns 4 consecutive m
#pragma unroll
            for (int it = 0; it < 4; ++it) {
                int kk = (tid >> 5) + it * 8;
                int mm = (tid & 31) * 4;
                float4 v = *(const float4*)&A[(long long)(k0 + kk) * lda + (m0 + mm)];
                As[mm + 0][kk] = (_Float16)v.x;
                As[mm + 1][kk] = (_Float16)v.y;
                As[mm + 2][kk] = (_Float16)v.z;
                As[mm + 3][kk] = (_Float16)v.w;
            }
        }

        // ---- stage B tile (BN x BK) ----
        if (TB == 0) {
            // memory contiguous along n: thread owns 4 consecutive n
#pragma unroll
            for (int it = 0; it < 2; ++it) {
                int kk = (tid >> 4) + it * 16;
                int nn = (tid & 15) * 4;
                float4 v = *(const float4*)&B[(long long)(k0 + kk) * ldb + (n0 + nn)];
                Bs[nn + 0][kk] = (_Float16)v.x;
                Bs[nn + 1][kk] = (_Float16)v.y;
                Bs[nn + 2][kk] = (_Float16)v.z;
                Bs[nn + 3][kk] = (_Float16)v.w;
            }
        } else {
            // memory contiguous along k: thread owns 4 consecutive k
#pragma unroll
            for (int it = 0; it < 2; ++it) {
                int nn = (tid >> 3) + it * 32;
                int kc = (tid & 7) * 4;
                float4 v = *(const float4*)&B[(long long)(n0 + nn) * ldb + (k0 + kc)];
                v4h h; h[0] = (_Float16)v.x; h[1] = (_Float16)v.y;
                       h[2] = (_Float16)v.z; h[3] = (_Float16)v.w;
                *(v4h*)&Bs[nn][kc] = h;   // single ds_store_b64
            }
        }
        __syncthreads();

        // A fragment (16x32, contiguous layout: lanes0-15 K{0..15}, lanes16-31 K{16..31})
        const int arow = wave * 16 + row16;
        const int klo  = half * 16;
        v8h alo = *(const v8h*)&As[arow][klo];
        v8h ahi = *(const v8h*)&As[arow][klo + 8];
        v16h afrag;
#pragma unroll
        for (int j = 0; j < 8; ++j) { afrag[j] = alo[j]; afrag[8 + j] = ahi[j]; }

#pragma unroll
        for (int tn = 0; tn < 4; ++tn) {
            // B fragment (32x16): lane n -> col, K 0..15 (lanes<16) / 16..31 (lanes>=16)
            const int bcol = tn * 16 + row16;
            v16h bfrag = *(const v16h*)&Bs[bcol][half * 16];
            acc[tn] = __builtin_amdgcn_wmma_f32_16x16x32_f16(
                false, afrag, false, bfrag, (short)0, acc[tn], false, false);
        }
        __syncthreads();
    }

    // epilogue: C layout lane L -> N = L&15, M = 8*(L>>4)+e  (exact tiling, no guards)
#pragma unroll
    for (int tn = 0; tn < 4; ++tn) {
#pragma unroll
        for (int e = 0; e < 8; ++e) {
            int m = m0 + wave * 16 + half * 8 + e;
            int n = n0 + tn * 16 + row16;
            float v = alpha * acc[tn][e];
            if (EPI == 0) {
                if (beta != 0.0f) v += beta * C[(long long)m * ldc + n];
                if (bias_mode == 1) v += bias[m];
                else if (bias_mode == 2) v += bias[n];
                C[(long long)m * ldc + n] = v;
            } else {
                // ConvTranspose2d k=2 s=2 scatter: m=(o,p,q), n over 32x32 input
                int o = m >> 2, p = (m >> 1) & 1, q = m & 1;
                int h = n >> 5, w = n & 31;
                if (bias) v += bias[o];
                C[(long long)o * HW2_ + (long long)(2 * h + p) * W2_ + (2 * w + q)] = v;
            }
        }
    }
}

// ---------------------------------------------------------------------------
// conv3x3 over concat([y, s2]) channels, pad=1, implicit-GEMM (im2col in B-stage)
// M=128 (o), N=4096 (h,w), K=2304 (c*9+tap); A = c1_w row-major [128][2304]
// ---------------------------------------------------------------------------
__global__ __launch_bounds__(256)
void conv3x3_wmma(const float* __restrict__ Wc, const float* __restrict__ yg,
                  const float* __restrict__ s2g, const float* __restrict__ bias,
                  float* __restrict__ zg)
{
    __shared__ __align__(32) _Float16 As[BM][BK];
    __shared__ __align__(32) _Float16 Bs[BN][BK];

    const int tid   = threadIdx.x;
    const int wave  = tid >> 5;
    const int lane  = tid & 31;
    const int row16 = lane & 15;
    const int half  = lane >> 4;

    const int n0 = blockIdx.x * BN;
    const long long zb = blockIdx.z;
    const float* y  = yg  + zb * (long long)OUTC_ * HW2_;
    const float* s2 = s2g + zb * (long long)OUTC_ * HW2_;
    float*       z  = zg  + zb * (long long)OUTC_ * HW2_;
    const int K = 2 * OUTC_ * 9;  // 2304

    v8f acc[4];
#pragma unroll
    for (int t = 0; t < 4; ++t)
#pragma unroll
        for (int e = 0; e < 8; ++e) acc[t][e] = 0.0f;

    for (int k0 = 0; k0 < K; k0 += BK) {
        // A: weights, contiguous along k -> float4 + ds_store_b64
#pragma unroll
        for (int it = 0; it < 4; ++it) {
            int mm = (tid >> 3) + it * 32;
            int kc = (tid & 7) * 4;
            float4 v = *(const float4*)&Wc[(long long)mm * K + (k0 + kc)];
            v4h h; h[0] = (_Float16)v.x; h[1] = (_Float16)v.y;
                   h[2] = (_Float16)v.z; h[3] = (_Float16)v.w;
            *(v4h*)&As[mm][kc] = h;
        }
        // B: im2col gather, lane-fast along n (consecutive w -> coalesced)
#pragma unroll
        for (int it = 0; it < 8; ++it) {
            int idx = tid + it * 256;
            int nn = idx & 63, kk = idx >> 6;
            int k = k0 + kk;
            int c = k / 9;
            int t9 = k - c * 9;
            int dy = t9 / 3;
            int dx = t9 - dy * 3;
            int n = n0 + nn;
            int h = (n >> 6) + dy - 1;
            int w = (n & 63) + dx - 1;
            float v = 0.0f;
            if ((unsigned)h < 64u && (unsigned)w < 64u) {
                const float* src = (c < OUTC_) ? y : s2;
                v = src[((long long)(c & (OUTC_ - 1))) * HW2_ + h * W2_ + w];
            }
            Bs[nn][kk] = (_Float16)v;
        }
        __syncthreads();

        const int arow = wave * 16 + row16;
        const int klo  = half * 16;
        v8h alo = *(const v8h*)&As[arow][klo];
        v8h ahi = *(const v8h*)&As[arow][klo + 8];
        v16h afrag;
#pragma unroll
        for (int j = 0; j < 8; ++j) { afrag[j] = alo[j]; afrag[8 + j] = ahi[j]; }

#pragma unroll
        for (int tn = 0; tn < 4; ++tn) {
            const int bcol = tn * 16 + row16;
            v16h bfrag = *(const v16h*)&Bs[bcol][half * 16];
            acc[tn] = __builtin_amdgcn_wmma_f32_16x16x32_f16(
                false, afrag, false, bfrag, (short)0, acc[tn], false, false);
        }
        __syncthreads();
    }

#pragma unroll
    for (int tn = 0; tn < 4; ++tn) {
#pragma unroll
        for (int e = 0; e < 8; ++e) {
            int m = wave * 16 + half * 8 + e;     // < 128
            int n = n0 + tn * 16 + row16;         // < 4096
            z[(long long)m * HW2_ + n] = acc[tn][e] + bias[m];
        }
    }
}

// ---------------------------------------------------------------------------
// softmax over rows of length 1024 (one block of 256 threads per row)
// ---------------------------------------------------------------------------
__global__ __launch_bounds__(256)
void softmax_rows_k(float* __restrict__ S)
{
    __shared__ float red[256];
    float* row = S + (long long)blockIdx.x * S_;
    const int c = threadIdx.x;
    float vals[4];
    float mx = -3.4e38f;
#pragma unroll
    for (int j = 0; j < 4; ++j) { vals[j] = row[c + j * 256]; mx = fmaxf(mx, vals[j]); }
    red[c] = mx; __syncthreads();
    for (int o = 128; o > 0; o >>= 1) {
        if (c < o) red[c] = fmaxf(red[c], red[c + o]);
        __syncthreads();
    }
    mx = red[0]; __syncthreads();
    float sm = 0.0f;
#pragma unroll
    for (int j = 0; j < 4; ++j) { vals[j] = __expf(vals[j] - mx); sm += vals[j]; }
    red[c] = sm; __syncthreads();
    for (int o = 128; o > 0; o >>= 1) {
        if (c < o) red[c] += red[c + o];
        __syncthreads();
    }
    float inv = 1.0f / red[0];
#pragma unroll
    for (int j = 0; j < 4; ++j) row[c + j * 256] = vals[j] * inv;
}

// ---------------------------------------------------------------------------
// LayerNorm(d=256) over rows of o2 [1024,256], then x[i][c][s] += result
// ---------------------------------------------------------------------------
__global__ __launch_bounds__(256)
void ln_add_k(const float* __restrict__ o2, const float* __restrict__ g,
              const float* __restrict__ b, float* __restrict__ xsl)
{
    __shared__ float red[256];
    const int s = blockIdx.x, c = threadIdx.x;
    float v = o2[(long long)s * INC_ + c];
    red[c] = v; __syncthreads();
    for (int o = 128; o > 0; o >>= 1) {
        if (c < o) red[c] += red[c + o];
        __syncthreads();
    }
    float mean = red[0] * (1.0f / 256.0f);
    __syncthreads();
    float d = v - mean;
    red[c] = d * d; __syncthreads();
    for (int o = 128; o > 0; o >>= 1) {
        if (c < o) red[c] += red[c + o];
        __syncthreads();
    }
    float var = red[0] * (1.0f / 256.0f);
    float out = d * rsqrtf(var + EPSF) * g[c] + b[c];
    xsl[(long long)c * S_ + s] += out;
}

// ---------------------------------------------------------------------------
// BatchNorm statistics per channel over (N,H,W): layout [64][128][4096]
// ---------------------------------------------------------------------------
__global__ __launch_bounds__(256)
void bn_stats_k(const float* __restrict__ t, float* __restrict__ mean,
                float* __restrict__ var)
{
    __shared__ float ssum[256];
    __shared__ float ssq[256];
    const int c = blockIdx.x;
    float s = 0.0f, q = 0.0f;
    for (int i = threadIdx.x; i < NN_ * HW2_; i += 256) {
        int n = i >> 12, hw = i & (HW2_ - 1);
        float v = t[((long long)n * OUTC_ + c) * HW2_ + hw];
        s += v; q += v * v;
    }
    ssum[threadIdx.x] = s; ssq[threadIdx.x] = q;
    __syncthreads();
    for (int o = 128; o > 0; o >>= 1) {
        if (threadIdx.x < o) {
            ssum[threadIdx.x] += ssum[threadIdx.x + o];
            ssq[threadIdx.x]  += ssq[threadIdx.x + o];
        }
        __syncthreads();
    }
    if (threadIdx.x == 0) {
        const float cnt = (float)(NN_ * HW2_);
        float m = ssum[0] / cnt;
        mean[c] = m;
        var[c]  = ssq[0] / cnt - m * m;
    }
}

// r = relu(bn(t1) + bn(t2)), written in place over t1
__global__ __launch_bounds__(256)
void bn_relu_combine_k(float* __restrict__ t1, const float* __restrict__ t2,
                       const float* __restrict__ st,
                       const float* __restrict__ g1, const float* __restrict__ b1,
                       const float* __restrict__ g2, const float* __restrict__ b2)
{
    long long idx = (long long)blockIdx.x * 256 + threadIdx.x;
    int c = (int)((idx >> 12) & (OUTC_ - 1));
    float a = (t1[idx] - st[c])       * rsqrtf(st[128 + c] + EPSF) * g1[c] + b1[c];
    float d = (t2[idx] - st[256 + c]) * rsqrtf(st[384 + c] + EPSF) * g2[c] + b2[c];
    t1[idx] = fmaxf(a + d, 0.0f);
}

// s2 = sigmoid(gate) * s
__global__ __launch_bounds__(256)
void sigmoid_mul_k(const float* __restrict__ gate, const float* __restrict__ s,
                   float* __restrict__ s2)
{
    long long idx = (long long)blockIdx.x * 256 + threadIdx.x;
    s2[idx] = s[idx] * (1.0f / (1.0f + __expf(-gate[idx])));
}

// out = relu(bn(z)) in place
__global__ __launch_bounds__(256)
void final_bn_relu_k(float* __restrict__ z, const float* __restrict__ mean,
                     const float* __restrict__ var, const float* __restrict__ g,
                     const float* __restrict__ b)
{
    long long idx = (long long)blockIdx.x * 256 + threadIdx.x;
    int c = (int)((idx >> 12) & (OUTC_ - 1));
    float v = (z[idx] - mean[c]) * rsqrtf(var[c] + EPSF) * g[c] + b[c];
    z[idx] = fmaxf(v, 0.0f);
}

// ---------------------------------------------------------------------------
extern "C" void kernel_launch(void* const* d_in, const int* in_sizes, int n_in,
                              void* d_out, int out_size, void* d_ws, size_t ws_size,
                              hipStream_t stream)
{
    (void)in_sizes; (void)n_in; (void)out_size; (void)ws_size;
    const float* x       = (const float*)d_in[0];
    const float* sIn     = (const float*)d_in[1];
    const float* wq      = (const float*)d_in[2];
    const float* bq      = (const float*)d_in[3];
    const float* wk      = (const float*)d_in[4];
    const float* bk      = (const float*)d_in[5];
    const float* wv      = (const float*)d_in[6];
    const float* bv      = (const float*)d_in[7];
    const float* wo      = (const float*)d_in[8];
    const float* bo      = (const float*)d_in[9];
    const float* ln_g    = (const float*)d_in[10];
    const float* ln_b    = (const float*)d_in[11];
    const float* up_w    = (const float*)d_in[12];
    const float* up_b    = (const float*)d_in[13];
    const float* wg_w    = (const float*)d_in[14];
    const float* wg_b    = (const float*)d_in[15];
    const float* wg_g    = (const float*)d_in[16];
    const float* wg_beta = (const float*)d_in[17];
    const float* ws_w    = (const float*)d_in[18];
    const float* ws_b    = (const float*)d_in[19];
    const float* ws_g    = (const float*)d_in[20];
    const float* ws_beta = (const float*)d_in[21];
    const float* out_w   = (const float*)d_in[22];
    const float* out_b   = (const float*)d_in[23];
    const float* c1_w    = (const float*)d_in[24];
    const float* c1_b    = (const float*)d_in[25];
    const float* c1_g    = (const float*)d_in[26];
    const float* c1_beta = (const float*)d_in[27];

    // workspace carve (floats)
    float* wsp = (float*)d_ws;
    size_t off = 0;
    const long long FR   = (long long)INC_ * S_;          // 256*1024 per frame
    const long long CHW2 = (long long)OUTC_ * HW2_;       // 128*4096 per batch
    float* xw  = wsp + off; off += (size_t)NN_ * FR;      // mutable copy of x
    float* qb  = wsp + off; off += (size_t)S_ * INC_;
    float* kb  = wsp + off; off += (size_t)2 * S_ * INC_;
    float* vb  = wsp + off; off += (size_t)2 * S_ * INC_;
    float* scb = wsp + off; off += (size_t)2 * S_ * S_;
    float* ob  = wsp + off; off += (size_t)S_ * INC_;
    float* o2b = wsp + off; off += (size_t)S_ * INC_;
    float* yb  = wsp + off; off += (size_t)NN_ * CHW2;
    float* t1b = wsp + off; off += (size_t)NN_ * CHW2;    // later reused as R
    float* t2b = wsp + off; off += (size_t)NN_ * CHW2;    // later reused as gate
    float* s2b = wsp + off; off += (size_t)NN_ * CHW2;
    float* stb = wsp + off; off += 1024;                  // bn stats

    float* zout = (float*)d_out;

    hipMemcpyAsync(xw, x, (size_t)NN_ * FR * sizeof(float),
                   hipMemcpyDeviceToDevice, stream);

    const float scale = 0.0625f;  // 1/sqrt(256)

    // ---------------- sequential rotary attention scan: i = 1..62 -----------
    for (int i = 1; i <= NN_ - 2; ++i) {
        const float* Xi = xw + (long long)i * FR;
        const float* Xm = xw + (long long)(i - 1) * FR;  // frames i-1, i+1 (sA=2*FR)

        // q = Xi^T @ wq + bq     [1024,256]
        gemm_wmma<1, 0, 0><<<dim3(INC_ / BN, S_ / BM, 1), 256, 0, stream>>>(
            Xi, wq, qb, bq, S_, INC_, INC_, S_, INC_, INC_,
            0, 0, 0, 1.0f, 0.0f, 2);
        // k,v for both context frames (batch 2)
        gemm_wmma<1, 0, 0><<<dim3(INC_ / BN, S_ / BM, 2), 256, 0, stream>>>(
            Xm, wk, kb, bk, S_, INC_, INC_, S_, INC_, INC_,
            2 * FR, 0, (long long)S_ * INC_, 1.0f, 0.0f, 2);
        gemm_wmma<1, 0, 0><<<dim3(INC_ / BN, S_ / BM, 2), 256, 0, stream>>>(
            Xm, wv, vb, bv, S_, INC_, INC_, S_, INC_, INC_,
            2 * FR, 0, (long long)S_ * INC_, 1.0f, 0.0f, 2);
        // scores = scale * q @ k^T   (batch 2)  [1024,1024]
        gemm_wmma<0, 1, 0><<<dim3(S_ / BN, S_ / BM, 2), 256, 0, stream>>>(
            qb, kb, scb, nullptr, S_, S_, INC_, INC_, INC_, S_,
            0, (long long)S_ * INC_, (long long)S_ * S_, scale, 0.0f, 0);
        softmax_rows_k<<<2 * S_, 256, 0, stream>>>(scb);
        // o = 0.5 * (a0@v0 + a1@v1)
        for (int f = 0; f < 2; ++f)
            gemm_wmma<0, 0, 0><<<dim3(INC_ / BN, S_ / BM, 1), 256, 0, stream>>>(
                scb + (long long)f * S_ * S_, vb + (long long)f * S_ * INC_, ob, nullptr,
                S_, INC_, S_, S_, INC_, INC_, 0, 0, 0, 0.5f, f ? 1.0f : 0.0f, 0);
        // o2 = o @ wo + bo
        gemm_wmma<0, 0, 0><<<dim3(INC_ / BN, S_ / BM, 1), 256, 0, stream>>>(
            ob, wo, o2b, bo, S_, INC_, INC_, INC_, INC_, INC_,
            0, 0, 0, 1.0f, 0.0f, 2);
        // x[i] += LN(o2)^T
        ln_add_k<<<S_, 256, 0, stream>>>(o2b, ln_g, ln_b, (float*)Xi);
    }

    // ---------------- ConvTranspose2d 2x2 s2: y = up(x) + up_b -------------
    gemm_wmma<1, 0, 1><<<dim3(S_ / BN, 512 / BM, NN_), 256, 0, stream>>>(
        up_w, xw, yb, up_b, 512, S_, INC_, 512, S_, 0,
        0, FR, CHW2, 1.0f, 0.0f, 0);

    // ---------------- attention gate ---------------------------------------
    // t1 = conv1x1(y; wg_w) + wg_b ; t2 = conv1x1(s; ws_w) + ws_b
    gemm_wmma<0, 0, 0><<<dim3(HW2_ / BN, OUTC_ / BM, NN_), 256, 0, stream>>>(
        wg_w, yb, t1b, wg_b, OUTC_, HW2_, OUTC_, OUTC_, HW2_, HW2_,
        0, CHW2, CHW2, 1.0f, 0.0f, 1);
    gemm_wmma<0, 0, 0><<<dim3(HW2_ / BN, OUTC_ / BM, NN_), 256, 0, stream>>>(
        ws_w, sIn, t2b, ws_b, OUTC_, HW2_, OUTC_, OUTC_, HW2_, HW2_,
        0, CHW2, CHW2, 1.0f, 0.0f, 1);
    bn_stats_k<<<OUTC_, 256, 0, stream>>>(t1b, stb + 0,   stb + 128);
    bn_stats_k<<<OUTC_, 256, 0, stream>>>(t2b, stb + 256, stb + 384);
    const long long TOT = (long long)NN_ * CHW2;  // 33.5M
    bn_relu_combine_k<<<(unsigned)(TOT / 256), 256, 0, stream>>>(
        t1b, t2b, stb, wg_g, wg_beta, ws_g, ws_beta);
    // gate = conv1x1(R; out_w) + out_b  (into t2b), then s2 = sigmoid(gate)*s
    gemm_wmma<0, 0, 0><<<dim3(HW2_ / BN, OUTC_ / BM, NN_), 256, 0, stream>>>(
        out_w, t1b, t2b, out_b, OUTC_, HW2_, OUTC_, OUTC_, HW2_, HW2_,
        0, CHW2, CHW2, 1.0f, 0.0f, 1);
    sigmoid_mul_k<<<(unsigned)(TOT / 256), 256, 0, stream>>>(t2b, sIn, s2b);

    // ---------------- conv3x3 over concat([y,s2]) + BN + ReLU --------------
    conv3x3_wmma<<<dim3(HW2_ / BN, 1, NN_), 256, 0, stream>>>(
        c1_w, yb, s2b, c1_b, zout);
    bn_stats_k<<<OUTC_, 256, 0, stream>>>(zout, stb + 512, stb + 640);
    final_bn_relu_k<<<(unsigned)(TOT / 256), 256, 0, stream>>>(
        zout, stb + 512, stb + 640, c1_g, c1_beta);
}
